// SpaMamba_47425028882459
// MI455X (gfx1250) — matrix operand order repaired and checked by reading, not compiled, not run on hardware
//
#include <hip/hip_runtime.h>
#include <hip/hip_bf16.h>
#include <math.h>

// ---------------------------------------------------------------------------
// Mamba block for MI455X (gfx1250, wave32, WMMA + TDM).
//   k_prep      : fp32 weights -> f16 packs, A = -exp(A_log)
//   k_u16       : x[B,C,H,W] -> u16[L,64] via LDS tile transpose
//   k_gemm_in   : xz = u @ in_proj^T (WMMA) -> xh16/z16, LDS-staged b128 stores
//   k_conv      : causal depthwise conv(4) + silu -> xc16
//   k_gemm_xp   : dbc = xc @ x_proj^T (WMMA, N 36->48) -> dbc, LDS-staged
//   k_delta     : delta = softplus(dt @ dt_proj^T + b)
//   k_scan1/mid/3 : chunked parallel linear scan, 1 d-channel per thread
//                   (16 states in registers), all streams coalesced
//   k_out       : TDM weights->LDS, out_proj GEMM -> proj GEMM -> LN -> silu -> +x
// ---------------------------------------------------------------------------

typedef _Float16 half8 __attribute__((ext_vector_type(8)));
typedef _Float16 v16h  __attribute__((ext_vector_type(16)));
typedef float    v8f   __attribute__((ext_vector_type(8)));
typedef unsigned int tdm_u4 __attribute__((ext_vector_type(4)));
typedef int          tdm_i4 __attribute__((ext_vector_type(4)));
typedef int          tdm_i8 __attribute__((ext_vector_type(8)));

#define L_SEQ   65536      // B*H*W
#define HW_SZ   16384      // H*W
#define DI      128        // d_inner
#define NDBC    48         // dt_rank(4) + 2*D_STATE(32), padded to 48
#define CHUNK   128
#define NCH     (L_SEQ / CHUNK)   // 512

// ---- WMMA fragment loaders (CDNA5 16-bit layouts, CDNA5 ISA §7.12.2) -----
__device__ __forceinline__ v16h load_a16(const _Float16* base, int ld, int k0, int lane) {
  const int r  = lane & 15;
  const int hi = (lane >> 4) & 1;
  const _Float16* p = base + (size_t)r * ld + k0 + hi * 8;
  half8 lo = *(const half8*)(p);
  half8 hh = *(const half8*)(p + 16);
  v16h f;
#pragma unroll
  for (int i = 0; i < 8; ++i) { f[i] = lo[i]; f[i + 8] = hh[i]; }
  return f;
}

__device__ __forceinline__ v16h load_b16(const _Float16* base, int ld, int k0, int lane) {
  const int r  = lane & 15;
  const int hi = (lane >> 4) & 1;
  const _Float16* p = base + (size_t)r * ld + k0 + hi * 8;
  half8 lo = *(const half8*)(p);
  half8 hh = *(const half8*)(p + 16);
  v16h f;
#pragma unroll
  for (int i = 0; i < 8; ++i) { f[i] = lo[i]; f[i + 8] = hh[i]; }
  return f;
}

__device__ __forceinline__ v8f wmma_f16(v16h a, v16h b, v8f c) {
  return __builtin_amdgcn_wmma_f32_16x16x32_f16(false, a, false, b, (short)0, c, false, false);
}

__device__ __forceinline__ float silu_f(float v) { return v / (1.f + __expf(-v)); }

// ---- TDM 2D tile load (descriptor per CDNA5 ISA) ---
// 2-byte elements; tile == full [rows x elems_per_row] block, row-major.
// This toolchain exposes the 6-arg builtin:
//   (uint32x4 g0, int32x8 g1, int32x4 g2, int32x4 g3, int32x8 pad, i32 cpol)
__device__ __forceinline__ void tdm_load_2d_f16(const void* gptr, unsigned lds_off,
                                                unsigned elems_per_row, unsigned rows,
                                                unsigned stride_elems) {
  const unsigned long long ga = (unsigned long long)gptr;
  tdm_u4 g0;
  g0[0] = 1u;                                              // count=1, user mode
  g0[1] = lds_off;                                         // LDS byte address
  g0[2] = (unsigned)(ga & 0xffffffffu);                    // global_addr[31:0]
  g0[3] = (unsigned)((ga >> 32) & 0x01ffffffu) | (2u << 30); // addr[56:32]|type=2
  tdm_i8 g1;
  g1[0] = (int)(1u << 16);                                 // data_size=1 (2B)
  g1[1] = (int)(elems_per_row << 16);                      // tensor_dim0[15:0]
  g1[2] = (int)(rows << 16);                               // dim0_hi=0 | tensor_dim1
  g1[3] = (int)(elems_per_row << 16);                      // dim1_hi=0 | tile_dim0
  g1[4] = (int)rows;                                       // tile_dim1 | tile_dim2=0
  g1[5] = (int)stride_elems;                               // tensor_dim0_stride lo
  g1[6] = 0;
  g1[7] = 0;
  tdm_i4 gz = {0, 0, 0, 0};                                // 2D: groups 2/3 unused
  tdm_i8 gz8 = {0, 0, 0, 0, 0, 0, 0, 0};
  __builtin_amdgcn_tensor_load_to_lds(g0, g1, gz, gz, gz8, 0);
}

// ---------------------------------------------------------------------------
__global__ void k_prep(const float* __restrict__ w_in_f, const float* __restrict__ w_xp_f,
                       const float* __restrict__ w_out_f, const float* __restrict__ w_pj_f,
                       const float* __restrict__ A_log,
                       _Float16* __restrict__ w_in, _Float16* __restrict__ w_xp,
                       _Float16* __restrict__ w_out, _Float16* __restrict__ w_pj,
                       float* __restrict__ Aval) {
  const int stride = gridDim.x * blockDim.x;
  const int tid = blockIdx.x * blockDim.x + threadIdx.x;
  for (int i = tid; i < 256 * 64; i += stride) w_in[i] = (_Float16)w_in_f[i];
  for (int i = tid; i < 48 * 128; i += stride) {
    int n = i >> 7, k = i & 127;
    w_xp[i] = (n < 36) ? (_Float16)w_xp_f[n * 128 + k] : (_Float16)0.f;
  }
  for (int i = tid; i < 64 * 128; i += stride) w_out[i] = (_Float16)w_out_f[i];
  for (int i = tid; i < 64 * 64;  i += stride) w_pj[i]  = (_Float16)w_pj_f[i];
  for (int i = tid; i < 128 * 16; i += stride) Aval[i]  = -__expf(A_log[i]);
}

// x[B,C,H,W] -> u16[L,64] via 64x64 LDS transpose (both sides coalesced)
__global__ void k_u16(const float* __restrict__ x, _Float16* __restrict__ u16) {
  __shared__ _Float16 tile[64][65];
  const int bid = blockIdx.x;                 // 1024 blocks = 4 * (16384/64)
  const int b   = bid >> 8;
  const int hw0 = (bid & 255) << 6;
  const int c0  = threadIdx.x >> 6;           // 0..3
  const int i   = threadIdx.x & 63;
#pragma unroll
  for (int s = 0; s < 16; ++s) {
    const int c = s * 4 + c0;
    tile[c][i] = (_Float16)x[(((size_t)b * 64 + c) << 14) + hw0 + i];
  }
  __syncthreads();
  const size_t t0 = (size_t)b * HW_SZ + hw0;
#pragma unroll
  for (int s = 0; s < 16; ++s) {
    const int r = s * 4 + c0;
    u16[(t0 + r) * 64 + i] = tile[i][r];
  }
}

// xz = u @ in_proj^T : 16 rows x 256 cols per block (4 waves x 4 n-tiles)
__global__ void k_gemm_in(const _Float16* __restrict__ u16, const _Float16* __restrict__ w16,
                          _Float16* __restrict__ xh16, _Float16* __restrict__ z16) {
  __shared__ __align__(16) _Float16 ot[16 * 256];
  __builtin_prefetch(w16, 0, 1);
  const int t0   = blockIdx.x * 16;
  const int wave = threadIdx.x >> 5;
  const int lane = threadIdx.x & 31;
  const _Float16* arow = u16 + (size_t)t0 * 64;
  v16h a0 = load_a16(arow, 64, 0,  lane);
  v16h a1 = load_a16(arow, 64, 32, lane);
  const int col = lane & 15, hi = lane >> 4;
#pragma unroll
  for (int nt = 0; nt < 4; ++nt) {
    const int n0 = wave * 64 + nt * 16;
    const _Float16* bbase = w16 + (size_t)n0 * 64;
    v16h b0 = load_b16(bbase, 64, 0,  lane);
    v16h b1 = load_b16(bbase, 64, 32, lane);
    v8f c = {};
    c = wmma_f16(a0, b0, c);
    c = wmma_f16(a1, b1, c);
#pragma unroll
    for (int v = 0; v < 8; ++v) ot[(v + hi * 8) * 256 + n0 + col] = (_Float16)c[v];
  }
  __syncthreads();
  // coalesced b128 copy-out, branchless base select (xh: segs 0..15, z: 16..31)
  for (int i = threadIdx.x; i < 16 * 32; i += 128) {
    const int row = i >> 5, seg = i & 31;
    const half8 v = *(const half8*)(ot + row * 256 + seg * 8);
    _Float16* dst = (seg < 16)
        ? (xh16 + (size_t)(t0 + row) * DI + seg * 8)
        : (z16  + (size_t)(t0 + row) * DI + (seg - 16) * 8);
    *(half8*)dst = v;
  }
}

// causal depthwise conv (kernel 4) over t, + silu
__global__ void k_conv(const _Float16* __restrict__ xh16, const float* __restrict__ conv_w,
                       const float* __restrict__ conv_b, _Float16* __restrict__ xc16) {
  const int idx = blockIdx.x * blockDim.x + threadIdx.x;   // L*128 total
  const int t = idx >> 7, d = idx & 127;
  float acc = conv_b[d];
#pragma unroll
  for (int k = 0; k < 4; ++k) {
    const int tt = t - 3 + k;
    if (tt >= 0) acc += (float)xh16[((size_t)tt << 7) + d] * conv_w[d * 4 + k];
  }
  xc16[idx] = (_Float16)silu_f(acc);
}

// dbc = xc @ x_proj^T (N=48): 64 rows per block, wave per 16-row strip
__global__ void k_gemm_xp(const _Float16* __restrict__ xc16, const _Float16* __restrict__ w16,
                          float* __restrict__ dbc) {
  __shared__ __align__(16) float od[64 * 48];
  __builtin_prefetch(w16, 0, 1);
  const int t0b   = blockIdx.x * 64;
  const int strip = threadIdx.x >> 5;
  const int lane  = threadIdx.x & 31;
  const int t0    = t0b + strip * 16;
  const _Float16* arow = xc16 + (size_t)t0 * DI;
  v16h a[4];
#pragma unroll
  for (int ks = 0; ks < 4; ++ks) a[ks] = load_a16(arow, DI, ks * 32, lane);
  const int col = lane & 15, hi = lane >> 4;
#pragma unroll
  for (int nt = 0; nt < 3; ++nt) {
    const int n0 = nt * 16;
    const _Float16* bbase = w16 + (size_t)n0 * DI;
    v8f c = {};
#pragma unroll
    for (int ks = 0; ks < 4; ++ks) c = wmma_f16(a[ks], load_b16(bbase, DI, ks * 32, lane), c);
#pragma unroll
    for (int v = 0; v < 8; ++v)
      od[(strip * 16 + v + hi * 8) * 48 + n0 + col] = c[v];
  }
  __syncthreads();
  for (int i = threadIdx.x; i < 64 * 12; i += 128) {       // float4 granules
    const int row = i / 12, seg = i % 12;
    *(float4*)(dbc + (size_t)(t0b + row) * NDBC + seg * 4) =
        *(const float4*)(od + row * 48 + seg * 4);
  }
}

// delta = softplus(dt @ dt_proj^T + b)
__global__ void k_delta(const float* __restrict__ dbc, const float* __restrict__ dtw,
                        const float* __restrict__ dtb, float* __restrict__ delta) {
  const int idx = blockIdx.x * blockDim.x + threadIdx.x;   // L*128
  const int t = idx >> 7, d = idx & 127;
  float acc = dtb[d];
  const float* r = dbc + (size_t)t * NDBC;
#pragma unroll
  for (int j = 0; j < 4; ++j) acc += r[j] * dtw[d * 4 + j];
  delta[idx] = (acc > 20.f) ? acc : log1pf(__expf(acc));
}

__device__ __forceinline__ void load16f(float* dst, const float* src) {
#pragma unroll
  for (int q = 0; q < 4; ++q) {
    const float4 v = *(const float4*)(src + q * 4);
    dst[q * 4 + 0] = v.x; dst[q * 4 + 1] = v.y; dst[q * 4 + 2] = v.z; dst[q * 4 + 3] = v.w;
  }
}

// phase 1: per-chunk summaries; one d per thread, 16 states in registers.
__global__ void k_scan1(const float* __restrict__ delta, const _Float16* __restrict__ xc16,
                        const float* __restrict__ dbc, const float* __restrict__ Aval,
                        float* __restrict__ hsum, float* __restrict__ aprod) {
  const int ch = blockIdx.x;
  const int d  = threadIdx.x;                               // 128 threads
  float Av[16];
  load16f(Av, Aval + d * 16);
  float h[16], ap[16];
#pragma unroll
  for (int j = 0; j < 16; ++j) { h[j] = 0.f; ap[j] = 1.f; }
  const int t0 = ch * CHUNK;
  for (int s = 0; s < CHUNK; ++s) {
    const size_t t = (size_t)(t0 + s);
    const float dl = delta[(t << 7) + d];
    const float bx = dl * (float)xc16[(t << 7) + d];
    float bm[16];
    load16f(bm, dbc + t * NDBC + 4);                        // B row (broadcast)
#pragma unroll
    for (int j = 0; j < 16; ++j) {
      const float a = __expf(dl * Av[j]);
      h[j] = a * h[j] + bx * bm[j];
      ap[j] *= a;
    }
  }
  float* hs = hsum  + (size_t)ch * 2048 + d * 16;
  float* as = aprod + (size_t)ch * 2048 + d * 16;
#pragma unroll
  for (int q = 0; q < 4; ++q) {
    *(float4*)(hs + q * 4) = make_float4(h[q*4], h[q*4+1], h[q*4+2], h[q*4+3]);
    *(float4*)(as + q * 4) = make_float4(ap[q*4], ap[q*4+1], ap[q*4+2], ap[q*4+3]);
  }
}

// middle: sequential scan over chunk summaries; carry[ch] = state entering ch
__global__ void k_scan_mid(const float4* __restrict__ hsum, const float4* __restrict__ aprod,
                           float4* __restrict__ carry) {
  const int tid = threadIdx.x;                              // 512 threads, 1 block
  float4 c = {0.f, 0.f, 0.f, 0.f};
  for (int ch = 0; ch < NCH; ++ch) {
    const float4 hs = hsum [ch * 512 + tid];
    const float4 ap = aprod[ch * 512 + tid];
    carry[ch * 512 + tid] = c;
    c.x = ap.x * c.x + hs.x;  c.y = ap.y * c.y + hs.y;
    c.z = ap.z * c.z + hs.z;  c.w = ap.w * c.w + hs.w;
  }
}

// phase 3: re-run chunk from carry-in; fuse y = (scan + xc*D)*silu(z)
__global__ void k_scan3(const float* __restrict__ delta, const _Float16* __restrict__ xc16,
                        const float* __restrict__ dbc, const float* __restrict__ Aval,
                        const float* __restrict__ carry, const float* __restrict__ Dvec,
                        const _Float16* __restrict__ z16, _Float16* __restrict__ y16) {
  const int ch = blockIdx.x;
  const int d  = threadIdx.x;                               // 128 threads
  float Av[16], h[16];
  load16f(Av, Aval + d * 16);
  load16f(h, carry + (size_t)ch * 2048 + d * 16);
  const float Dv = Dvec[d];
  const int t0 = ch * CHUNK;
  for (int s = 0; s < CHUNK; ++s) {
    const size_t t = (size_t)(t0 + s);
    const float dl = delta[(t << 7) + d];
    const float xv = (float)xc16[(t << 7) + d];
    const float bx = dl * xv;
    float bm[16], cm[16];
    load16f(bm, dbc + t * NDBC + 4);
    load16f(cm, dbc + t * NDBC + 20);
    float ps = 0.f;
#pragma unroll
    for (int j = 0; j < 16; ++j) {
      const float a = __expf(dl * Av[j]);
      h[j] = a * h[j] + bx * bm[j];
      ps += h[j] * cm[j];
    }
    const float zf = (float)z16[(t << 7) + d];
    y16[(t << 7) + d] = (_Float16)((ps + xv * Dv) * silu_f(zf));
  }
}

// fused: t1 = y @ out_proj^T ; p = t1 @ proj^T + b ; LN ; silu ; + x
// Weights staged into LDS by the Tensor Data Mover.
__global__ void k_out(const _Float16* __restrict__ y16, const _Float16* __restrict__ wo16,
                      const _Float16* __restrict__ wp16, const float* __restrict__ proj_b,
                      const float* __restrict__ ln_g, const float* __restrict__ ln_b,
                      const float* __restrict__ x, float* __restrict__ out) {
  __shared__ __align__(16) _Float16 s_wo[64 * 128];
  __shared__ __align__(16) _Float16 s_wp[64 * 64];
  __shared__ __align__(16) _Float16 lt1[16 * 64];
  __shared__ float lp[16 * 64];
  const int t0 = blockIdx.x * 16;
  const int lane = threadIdx.x;                             // 32 threads = 1 wave

  // TDM: DMA both weight matrices into LDS (one wave -> issued once each)
  tdm_load_2d_f16(wo16, (unsigned)(unsigned long long)(uintptr_t)s_wo, 128u, 64u, 128u);
  tdm_load_2d_f16(wp16, (unsigned)(unsigned long long)(uintptr_t)s_wp, 64u,  64u, 64u);

  const _Float16* arow = y16 + (size_t)t0 * DI;
  v16h a[4];
#pragma unroll
  for (int ks = 0; ks < 4; ++ks) a[ks] = load_a16(arow, DI, ks * 32, lane);
  const int col = lane & 15, hi = lane >> 4;

  __builtin_amdgcn_s_wait_tensorcnt(0);
#pragma unroll
  for (int nt = 0; nt < 4; ++nt) {
    v8f c = {};
#pragma unroll
    for (int ks = 0; ks < 4; ++ks)
      c = wmma_f16(a[ks], load_b16(s_wo + (size_t)(nt * 16) * DI, DI, ks * 32, lane), c);
#pragma unroll
    for (int v = 0; v < 8; ++v) lt1[(v + hi * 8) * 64 + nt * 16 + col] = (_Float16)c[v];
  }
  __syncthreads();
  v16h a2[2];
#pragma unroll
  for (int ks = 0; ks < 2; ++ks) a2[ks] = load_a16(lt1, 64, ks * 32, lane);
#pragma unroll
  for (int nt = 0; nt < 4; ++nt) {
    v8f c = {};
#pragma unroll
    for (int ks = 0; ks < 2; ++ks)
      c = wmma_f16(a2[ks], load_b16(s_wp + (size_t)(nt * 16) * 64, 64, ks * 32, lane), c);
#pragma unroll
    for (int v = 0; v < 8; ++v)
      lp[(v + hi * 8) * 64 + nt * 16 + col] = c[v] + proj_b[nt * 16 + col];
  }
  __syncthreads();
  if (lane < 16) {                                          // one row per lane
    const int t = t0 + lane;
    const int b = t >> 14, hw = t & (HW_SZ - 1);
    float mu = 0.f;
    for (int c = 0; c < 64; ++c) mu += lp[lane * 64 + c];
    mu *= (1.f / 64.f);
    float var = 0.f;
    for (int c = 0; c < 64; ++c) { const float dv = lp[lane * 64 + c] - mu; var += dv * dv; }
    const float rs = rsqrtf(var * (1.f / 64.f) + 1e-5f);
    for (int c = 0; c < 64; ++c) {
      const float pv = (lp[lane * 64 + c] - mu) * rs * ln_g[c] + ln_b[c];
      const size_t xi = (((size_t)b * 64 + c) << 14) + hw;  // coalesced across lanes
      out[xi] = silu_f(pv) + x[xi];
    }
  }
}

// ---------------------------------------------------------------------------
extern "C" void kernel_launch(void* const* d_in, const int* in_sizes, int n_in,
                              void* d_out, int out_size, void* d_ws, size_t ws_size,
                              hipStream_t stream) {
  (void)in_sizes; (void)n_in; (void)out_size; (void)ws_size;
  const float* x        = (const float*)d_in[0];
  const float* in_proj  = (const float*)d_in[1];
  const float* conv_w   = (const float*)d_in[2];
  const float* conv_b   = (const float*)d_in[3];
  const float* x_proj   = (const float*)d_in[4];
  const float* dt_w     = (const float*)d_in[5];
  const float* dt_b     = (const float*)d_in[6];
  const float* A_log    = (const float*)d_in[7];
  const float* Dvec     = (const float*)d_in[8];
  const float* out_proj = (const float*)d_in[9];
  const float* proj_w   = (const float*)d_in[10];
  const float* proj_b   = (const float*)d_in[11];
  const float* ln_g     = (const float*)d_in[12];
  const float* ln_b     = (const float*)d_in[13];
  float* out = (float*)d_out;

  char* ws = (char*)d_ws;
  size_t off = 0;
  auto take = [&](size_t bytes) {
    off = (off + 255) & ~(size_t)255;
    char* p = ws + off; off += bytes; return p;
  };
  _Float16* w_in  = (_Float16*)take(256 * 64 * 2);
  _Float16* w_xp  = (_Float16*)take(48 * 128 * 2);
  _Float16* w_out = (_Float16*)take(64 * 128 * 2);
  _Float16* w_pj  = (_Float16*)take(64 * 64 * 2);
  float*    Aval  = (float*)   take(128 * 16 * 4);
  _Float16* u16   = (_Float16*)take((size_t)L_SEQ * 64  * 2);
  _Float16* xh16  = (_Float16*)take((size_t)L_SEQ * DI  * 2);
  _Float16* z16   = (_Float16*)take((size_t)L_SEQ * DI  * 2);
  _Float16* xc16  = (_Float16*)take((size_t)L_SEQ * DI  * 2);
  float*    dbc   = (float*)   take((size_t)L_SEQ * NDBC * 4);
  float*    delta = (float*)   take((size_t)L_SEQ * DI  * 4);
  _Float16* y16   = (_Float16*)take((size_t)L_SEQ * DI  * 2);
  float*    hsum  = (float*)   take((size_t)NCH * 2048 * 4);
  float*    aprod = (float*)   take((size_t)NCH * 2048 * 4);
  float*    carry = (float*)   take((size_t)NCH * 2048 * 4);

  k_prep<<<64, 256, 0, stream>>>(in_proj, x_proj, out_proj, proj_w, A_log,
                                 w_in, w_xp, w_out, w_pj, Aval);
  k_u16<<<L_SEQ / 64, 256, 0, stream>>>(x, u16);
  k_gemm_in<<<L_SEQ / 16, 128, 0, stream>>>(u16, w_in, xh16, z16);
  k_conv<<<(L_SEQ * DI) / 256, 256, 0, stream>>>(xh16, conv_w, conv_b, xc16);
  k_gemm_xp<<<L_SEQ / 64, 128, 0, stream>>>(xc16, w_xp, dbc);
  k_delta<<<(L_SEQ * DI) / 256, 256, 0, stream>>>(dbc, dt_w, dt_b, delta);
  k_scan1<<<NCH, 128, 0, stream>>>(delta, xc16, dbc, Aval, hsum, aprod);
  k_scan_mid<<<1, 512, 0, stream>>>((const float4*)hsum, (const float4*)aprod, (float4*)carry);
  k_scan3<<<NCH, 128, 0, stream>>>(delta, xc16, dbc, Aval, carry, Dvec, z16, y16);
  k_out<<<L_SEQ / 16, 32, 0, stream>>>(y16, w_out, w_pj, proj_b, ln_g, ln_b, x, out);
}
